// DiffusionBends_82841329205438
// MI455X (gfx1250) — matrix operand order repaired and checked by reading, not compiled, not run on hardware
//
#include <hip/hip_runtime.h>

// ---------------------------------------------------------------------------
// DiffusionBends on MI455X (gfx1250): bf16 WMMA MLP, LDS-staged weights/feats
// ---------------------------------------------------------------------------

typedef __attribute__((ext_vector_type(16))) __bf16 v16bf;
typedef __attribute__((ext_vector_type(8)))  float  v8f;

#define NWAVES 4
#define BLOCK  (NWAVES * 32)

__device__ __forceinline__ unsigned short f2bf(float f) {
    // round-to-nearest-even float -> bf16
    unsigned int u = __float_as_uint(f);
    u += 0x7FFFu + ((u >> 16) & 1u);
    return (unsigned short)(u >> 16);
}

union FragU {
    uint4 u[2];   // 32 bytes
    v16bf v;      // 16 x bf16
};

__global__ __launch_bounds__(256) void copy_init_kernel(const float* __restrict__ src,
                                                        float* __restrict__ dst, int n) {
    int i = blockIdx.x * blockDim.x + threadIdx.x;
    if (i < n) dst[i] = src[i];
}

__global__ __launch_bounds__(BLOCK) void bends_kernel(
    const float* __restrict__ coords,   // [N_ATOMS][8][3]
    const int*   __restrict__ angles,   // [A][3]
    const int*   __restrict__ rings,    // [A]
    const float* __restrict__ encoded,  // [N_ATOMS][64]
    const float* __restrict__ tvals,    // [8]
    const float* __restrict__ ring_emb, // [128][10]
    const float* __restrict__ W0, const float* __restrict__ b0,   // [204][64],[64]
    const float* __restrict__ W1, const float* __restrict__ b1,   // [64][64],[64]
    const float* __restrict__ W2, const float* __restrict__ b2,   // [64][64],[64]
    const float* __restrict__ W3, const float* __restrict__ b3,   // [64][2],[2]
    float* __restrict__ out, int A)
{
    // Transposed bf16 weights: WT[n][k]; strides multiple of 8 halfwords (16B align)
    __shared__ alignas(16) unsigned short sWT0[64][232];   // K padded 204 -> 224
    __shared__ alignas(16) unsigned short sWT1[64][72];
    __shared__ alignas(16) unsigned short sWT2[64][72];
    __shared__ alignas(16) unsigned short sWT3[16][72];    // N padded 2 -> 16
    __shared__ alignas(16) unsigned short sFeat[NWAVES][2][232]; // per-angle base feats
    __shared__ alignas(16) unsigned short sH[NWAVES][16][72];    // inter-layer 16x64 tile
    __shared__ float sDelta[NWAVES][16][2];
    __shared__ float sB0[64], sB1[64], sB2[64], sB3[16];

    const int tid = threadIdx.x;

    // ---- stage weights (transpose + bf16 convert), zero padding ----
    for (int i = tid; i < 64 * 232; i += BLOCK) {
        int n = i / 232, k = i % 232;
        sWT0[n][k] = f2bf((k < 204) ? W0[k * 64 + n] : 0.0f);
    }
    for (int i = tid; i < 64 * 72; i += BLOCK) {
        int n = i / 72, k = i % 72;
        float v1 = (k < 64) ? W1[k * 64 + n] : 0.0f;
        float v2 = (k < 64) ? W2[k * 64 + n] : 0.0f;
        sWT1[n][k] = f2bf(v1);
        sWT2[n][k] = f2bf(v2);
    }
    for (int i = tid; i < 16 * 72; i += BLOCK) {
        int n = i / 72, k = i % 72;
        sWT3[n][k] = f2bf((n < 2 && k < 64) ? W3[k * 2 + n] : 0.0f);
    }
    for (int i = tid; i < 64; i += BLOCK) { sB0[i] = b0[i]; sB1[i] = b1[i]; sB2[i] = b2[i]; }
    if (tid < 16) sB3[tid] = (tid < 2) ? b3[tid] : 0.0f;

    const int wave = tid >> 5, lane = tid & 31;
    const int m = lane & 15;          // row within 16-row tile
    const int g = lane >> 4;          // lane group (0: lanes 0-15, 1: lanes 16-31)
    const int pairIdx   = blockIdx.x * NWAVES + wave;  // each wave: 2 angles x 8 t
    const int angleBase = pairIdx * 2;
    const int rowAngle  = angleBase + (m >> 3);
    const bool valid    = (rowAngle < A);
    const int aIdx      = valid ? rowAngle : 0;

    const int a0 = angles[aIdx * 3 + 0];
    const int a2 = angles[aIdx * 3 + 2];
    const int ti = m & 7;

    // geometry for my row (computed redundantly by lane m and lane m+16)
    const float drx = coords[(a0 * 8 + ti) * 3 + 0] - coords[(a2 * 8 + ti) * 3 + 0];
    const float dry = coords[(a0 * 8 + ti) * 3 + 1] - coords[(a2 * 8 + ti) * 3 + 1];
    const float drz = coords[(a0 * 8 + ti) * 3 + 2] - coords[(a2 * 8 + ti) * 3 + 2];
    const float dl  = sqrtf(fmaxf(drx * drx + dry * dry + drz * drz, 1e-12f));
    const float inv = 1.0f / dl;
    const float dhx = drx * inv, dhy = dry * inv, dhz = drz * inv;
    const float tv  = tvals[ti];

    // ---- stage per-angle base feature rows (shared across the 8 timesteps) ----
    // layout per angle: [0:64)=enc[a0], [64:128)=enc[a1], [128:192)=enc[a2],
    //                   [192:202)=ring*RING_NORM, [202:232)=0 (202/203 patched later)
    for (int i = lane; i < 2 * 232; i += 32) {
        int arow = i / 232, k = i % 232;
        int ang = angleBase + arow; if (ang >= A) ang = 0;
        float v = 0.0f;
        if (k < 192) {
            int atom = angles[ang * 3 + (k >> 6)];
            v = encoded[atom * 64 + (k & 63)];
        } else if (k < 202) {
            v = ring_emb[rings[ang] * 10 + (k - 192)] * 0.31622776601683794f; // 1/sqrt(10)
        }
        sFeat[wave][arow][k] = f2bf(v);
    }
    __syncthreads();

    // per-row features t (K=202) and dl (K=203): halves 2,3 of lanes 16-31, k-chunk 6
    const unsigned patch = ((unsigned)f2bf(dl) << 16) | (unsigned)f2bf(tv);
    const int arow_m = m >> 3;
    const v8f vzero = {0.f, 0.f, 0.f, 0.f, 0.f, 0.f, 0.f, 0.f};

    // ---- layer 0: feat(16x224) @ W0 -> h(16x64), 7 K-chunks x 4 N-tiles ----
    v8f acc[4];
    acc[0] = vzero; acc[1] = vzero; acc[2] = vzero; acc[3] = vzero;
    #pragma unroll
    for (int kc = 0; kc < 7; ++kc) {
        FragU a;
        const unsigned short* ap = &sFeat[wave][arow_m][kc * 32 + g * 8];
        a.u[0] = *(const uint4*)ap;            // halves 0-7 : K = kc*32 + g*8 + 0..7
        a.u[1] = *(const uint4*)(ap + 16);     // halves 8-15: K = kc*32 + 16 + g*8 + 0..7
        if (kc == 6) a.u[0].y = (g == 1) ? patch : a.u[0].y;  // K=202(t),203(dl)
        #pragma unroll
        for (int nt = 0; nt < 4; ++nt) {
            FragU b;   // lane = column N, K = kc*32 + g*8 + 0..7 and kc*32 + 16 + g*8 + 0..7
            const unsigned short* bp = &sWT0[nt * 16 + m][kc * 32 + g * 8];
            b.u[0] = *(const uint4*)bp;
            b.u[1] = *(const uint4*)(bp + 16);
            acc[nt] = __builtin_amdgcn_wmma_f32_16x16x32_bf16(
                false, a.v, false, b.v, (short)0, acc[nt], false, false);
        }
    }
    // epilogue: bias + leaky relu -> bf16 tile in LDS (row = g*8+r, col = nt*16+m)
    #pragma unroll
    for (int nt = 0; nt < 4; ++nt) {
        float bias = sB0[nt * 16 + m];
        #pragma unroll
        for (int r = 0; r < 8; ++r) {
            float x = acc[nt][r] + bias;
            x = (x >= 0.f) ? x : 0.001f * x;
            sH[wave][g * 8 + r][nt * 16 + m] = f2bf(x);
        }
    }
    __syncthreads();

    // ---- layers 1 and 2: h(16x64) @ W -> 16x64 ----
    for (int layer = 0; layer < 2; ++layer) {
        const unsigned short (*WT)[72] = layer ? sWT2 : sWT1;
        const float* bs = layer ? sB2 : sB1;
        v8f hacc[4];
        hacc[0] = vzero; hacc[1] = vzero; hacc[2] = vzero; hacc[3] = vzero;
        #pragma unroll
        for (int kc = 0; kc < 2; ++kc) {
            FragU a;
            const unsigned short* ap = &sH[wave][m][kc * 32 + g * 8];
            a.u[0] = *(const uint4*)ap;
            a.u[1] = *(const uint4*)(ap + 16);
            #pragma unroll
            for (int nt = 0; nt < 4; ++nt) {
                FragU b;
                const unsigned short* bp = &WT[nt * 16 + m][kc * 32 + g * 8];
                b.u[0] = *(const uint4*)bp;
                b.u[1] = *(const uint4*)(bp + 16);
                hacc[nt] = __builtin_amdgcn_wmma_f32_16x16x32_bf16(
                    false, a.v, false, b.v, (short)0, hacc[nt], false, false);
            }
        }
        __syncthreads();   // all reads of sH done before overwrite
        #pragma unroll
        for (int nt = 0; nt < 4; ++nt) {
            float bias = bs[nt * 16 + m];
            #pragma unroll
            for (int r = 0; r < 8; ++r) {
                float x = hacc[nt][r] + bias;
                x = (x >= 0.f) ? x : 0.001f * x;
                sH[wave][g * 8 + r][nt * 16 + m] = f2bf(x);
            }
        }
        __syncthreads();
    }

    // ---- layer 3: h(16x64) @ W3 (N padded to 16) -> delta in columns 0,1 ----
    v8f dacc = vzero;
    #pragma unroll
    for (int kc = 0; kc < 2; ++kc) {
        FragU a;
        const unsigned short* ap = &sH[wave][m][kc * 32 + g * 8];
        a.u[0] = *(const uint4*)ap;
        a.u[1] = *(const uint4*)(ap + 16);
        FragU b;
        const unsigned short* bp = &sWT3[m][kc * 32 + g * 8];
        b.u[0] = *(const uint4*)bp;
        b.u[1] = *(const uint4*)(bp + 16);
        dacc = __builtin_amdgcn_wmma_f32_16x16x32_bf16(
            false, a.v, false, b.v, (short)0, dacc, false, false);
    }
    if (m < 2) {
        float bias = sB3[m];
        #pragma unroll
        for (int r = 0; r < 8; ++r)
            sDelta[wave][g * 8 + r][m] = dacc[r] + bias;
    }
    __syncthreads();

    // ---- scatter: lanes 0-15 -> a0 contribution, lanes 16-31 -> a2 ----
    const float d0 = sDelta[wave][m][0];
    const float d1 = sDelta[wave][m][1];
    const float coef = (g == 0) ? (-0.5f * d0) : (0.5f * d1);
    const int atom = (g == 0) ? a0 : a2;
    if (valid) {
        atomicAdd(&out[(atom * 8 + ti) * 3 + 0], coef * dhx);
        atomicAdd(&out[(atom * 8 + ti) * 3 + 1], coef * dhy);
        atomicAdd(&out[(atom * 8 + ti) * 3 + 2], coef * dhz);
    }
}

extern "C" void kernel_launch(void* const* d_in, const int* in_sizes, int n_in,
                              void* d_out, int out_size, void* d_ws, size_t ws_size,
                              hipStream_t stream)
{
    const float* coords   = (const float*)d_in[0];
    const int*   angles   = (const int*)  d_in[1];
    const int*   rings    = (const int*)  d_in[2];
    const float* encoded  = (const float*)d_in[3];
    const float* t        = (const float*)d_in[4];
    const float* answer   = (const float*)d_in[5];
    const float* ring_emb = (const float*)d_in[6];
    const float* W0 = (const float*)d_in[7];
    const float* b0 = (const float*)d_in[8];
    const float* W1 = (const float*)d_in[9];
    const float* b1 = (const float*)d_in[10];
    const float* W2 = (const float*)d_in[11];
    const float* b2 = (const float*)d_in[12];
    const float* W3 = (const float*)d_in[13];
    const float* b3 = (const float*)d_in[14];
    float* out = (float*)d_out;

    const int A = in_sizes[1] / 3;

    // out starts as `answer` (reference accumulates into it)
    copy_init_kernel<<<(out_size + 255) / 256, 256, 0, stream>>>(answer, out, out_size);

    const int nPairs  = (A + 1) / 2;                    // 2 angles (16 rows) per wave
    const int nBlocks = (nPairs + NWAVES - 1) / NWAVES;
    bends_kernel<<<nBlocks, BLOCK, 0, stream>>>(
        coords, angles, rings, encoded, t, ring_emb,
        W0, b0, W1, b1, W2, b2, W3, b3, out, A);
}
